// LocalMHA_78726750536342
// MI455X (gfx1250) — matrix operand order repaired and checked by reading, not compiled, not run on hardware
//
#include <hip/hip_runtime.h>

// ---------------- CDNA5 WMMA / TDM types ----------------
typedef __attribute__((ext_vector_type(16))) __bf16 v16bf;
typedef __attribute__((ext_vector_type(8)))  __bf16 bf16x8;
typedef __attribute__((ext_vector_type(16))) __bf16 bf16x16;
typedef __attribute__((ext_vector_type(8)))  float  v8f;
typedef __attribute__((ext_vector_type(4)))  unsigned int u32x4;
typedef __attribute__((ext_vector_type(8)))  int  i32x8;
typedef __attribute__((ext_vector_type(4)))  int  i32x4;

#define NEGINF (-__builtin_inff())

__device__ __forceinline__ v8f wmma_bf16(v16bf a, v16bf b, v8f c) {
  // D = A(16x32 bf16) * B(32x16 bf16) + C(16x16 f32)
  return __builtin_amdgcn_wmma_f32_16x16x32_bf16(
      /*neg_a=*/false, a, /*neg_b=*/false, b,
      /*c_mod=*/(short)0, c, /*reuse_a=*/false, /*reuse_b=*/false);
}

// A-frag: lane = M row (lane&15); hi half selects K pattern:
//   elems 0..7  = K hi*8+0..7, elems 8..15 = K 16+hi*8+0..7
__device__ __forceinline__ v16bf load_a_frag(const __bf16* rowp, int hi) {
  union { v16bf f; bf16x8 h[2]; } u;
  u.h[0] = *reinterpret_cast<const bf16x8*>(rowp + hi * 8);
  u.h[1] = *reinterpret_cast<const bf16x8*>(rowp + 16 + hi * 8);
  return u.f;
}

// B-frag: lane = N col (lane&15); elems 0..7 = K hi*8+0..7, elems 8..15 = K 16+hi*8+0..7
__device__ __forceinline__ v16bf load_b_frag(const __bf16* colp, int hi) {
  union { v16bf f; bf16x8 h[2]; } u;
  u.h[0] = *reinterpret_cast<const bf16x8*>(colp + hi * 8);
  u.h[1] = *reinterpret_cast<const bf16x8*>(colp + 16 + hi * 8);
  return u.f;
}

__device__ __forceinline__ float hmax16(float x) {
  x = fmaxf(x, __shfl_xor(x, 1));
  x = fmaxf(x, __shfl_xor(x, 2));
  x = fmaxf(x, __shfl_xor(x, 4));
  x = fmaxf(x, __shfl_xor(x, 8));
  return x;
}
__device__ __forceinline__ float hsum16(float x) {
  x += __shfl_xor(x, 1);
  x += __shfl_xor(x, 2);
  x += __shfl_xor(x, 4);
  x += __shfl_xor(x, 8);
  return x;
}

// Low 32 bits of a generic pointer into LDS == LDS byte offset
__device__ __forceinline__ unsigned lds_off(const void* p) {
  return (unsigned)(unsigned long long)(uintptr_t)p;
}

// ---- Tensor Data Mover: 2D bf16 tile (tile_w x tile_h) global -> LDS ----
// D# group0: count=1 | lds_addr | global_addr(57b) | type=2
// D# group1: data_size=1(2B), tensor_dim0/1, tile_dim0/1, tensor_dim0_stride
// Groups 2/3 zero (2-D tensor); 5th operand is the clang-23 extra descriptor
// group (unused here), cpol=0.
__device__ __forceinline__ void tdm_load_2d_bf16(const __bf16* gptr, unsigned ldsoff,
                                                 unsigned tensor_w, unsigned tensor_h,
                                                 unsigned stride_elems,
                                                 unsigned tile_w, unsigned tile_h) {
  const unsigned long long ga = (unsigned long long)(uintptr_t)gptr;
  u32x4 g0;
  g0[0] = 1u;                                             // count=1, user mode
  g0[1] = ldsoff;                                         // lds_addr (bytes)
  g0[2] = (unsigned)(ga & 0xffffffffu);                   // global_addr[31:0]
  g0[3] = (unsigned)((ga >> 32) & 0x01ffffffu)            // global_addr[56:32]
          | 0x80000000u;                                  // type=2 ("image")
  i32x8 g1;
  g1[0] = 0x00010000;                                     // data_size=1 (2 bytes)
  g1[1] = (int)((tensor_w & 0xffffu) << 16);              // tensor_dim0[15:0]
  g1[2] = (int)((tensor_w >> 16) | ((tensor_h & 0xffffu) << 16));
  g1[3] = (int)((tensor_h >> 16) | (tile_w << 16));       // tile_dim0
  g1[4] = (int)(tile_h & 0xffffu);                        // tile_dim1 (tile_dim2=0)
  g1[5] = (int)stride_elems;                              // tensor_dim0_stride[31:0]
  g1[6] = 0;
  g1[7] = 0;
  const i32x4 z4 = {0, 0, 0, 0};                          // groups 2/3 unused (2-D)
  const i32x8 z8 = {0, 0, 0, 0, 0, 0, 0, 0};
  __builtin_amdgcn_tensor_load_to_lds(g0, g1, z4, z4, z8, 0);
}

// ---------------- fp32 -> bf16 convert ----------------
__global__ __launch_bounds__(256) void cvt_bf16_kernel(const float* __restrict__ src,
                                                       __bf16* __restrict__ dst, long n) {
  long i = (long)blockIdx.x * blockDim.x + threadIdx.x;
  long stride = (long)gridDim.x * blockDim.x;
  for (; i < n; i += stride) dst[i] = (__bf16)src[i];
}

// ---------------- LayerNorm: one block per token row (DIM=1024) ----------------
__global__ __launch_bounds__(256) void ln_kernel(const float* __restrict__ x,
                                                 const float* __restrict__ g,
                                                 const float* __restrict__ b,
                                                 __bf16* __restrict__ xn) {
  __shared__ float rs[256], rq[256];
  const long row = blockIdx.x;
  const int t = threadIdx.x;
  const float4 v4 = reinterpret_cast<const float4*>(x + row * 1024)[t];
  float s1 = v4.x + v4.y + v4.z + v4.w;
  float s2 = v4.x * v4.x + v4.y * v4.y + v4.z * v4.z + v4.w * v4.w;
  rs[t] = s1; rq[t] = s2;
  __syncthreads();
  for (int off = 128; off > 0; off >>= 1) {
    if (t < off) { rs[t] += rs[t + off]; rq[t] += rq[t + off]; }
    __syncthreads();
  }
  const float mu = rs[0] * (1.0f / 1024.0f);
  const float var = rq[0] * (1.0f / 1024.0f) - mu * mu;
  const float rstd = rsqrtf(var + 1e-5f);
  const float4 g4 = reinterpret_cast<const float4*>(g)[t];
  const float4 b4 = reinterpret_cast<const float4*>(b)[t];
  __bf16* o = xn + row * 1024 + t * 4;
  o[0] = (__bf16)((v4.x - mu) * rstd * g4.x + b4.x);
  o[1] = (__bf16)((v4.y - mu) * rstd * g4.y + b4.y);
  o[2] = (__bf16)((v4.z - mu) * rstd * g4.z + b4.z);
  o[3] = (__bf16)((v4.w - mu) * rstd * g4.w + b4.w);
}

// ---------------- GEMM: C[M,N] = A[M,K] @ B[N,K]^T  (bf16 in, f32 acc) ------
// block = 256 thr (8 waves, 2x4), wave tile 32x32, block tile 64x128.
// A/B tiles staged into LDS by the Tensor Data Mover, double-buffered:
// wave0 issues tensor_load_to_lds for chunk s+1, all waves compute chunk s
// from LDS, wave0 drains TENSORcnt, barrier.
template <bool OUT_F32>
__global__ __launch_bounds__(256) void gemm_bf16_kernel(const __bf16* __restrict__ A,
                                                        const __bf16* __restrict__ B,
                                                        void* __restrict__ Cout,
                                                        int M, int N, int K) {
  __shared__ __bf16 As[2][64 * 64];    //  8 KB x2
  __shared__ __bf16 Bs[2][128 * 64];   // 16 KB x2
  const int lane = threadIdx.x & 31;
  const int wave = threadIdx.x >> 5;
  const int r = lane & 15, hi = lane >> 4;
  const int wm = wave & 1, wn = wave >> 1;
  const long arow = (long)blockIdx.x * 64;
  const long bcol = (long)blockIdx.y * 128;

  v8f acc[2][2];
  const v8f vz = {0.f, 0.f, 0.f, 0.f, 0.f, 0.f, 0.f, 0.f};
  acc[0][0] = vz; acc[0][1] = vz; acc[1][0] = vz; acc[1][1] = vz;

  const int nchunks = K >> 6;  // K / 64
  if (wave == 0) {
    tdm_load_2d_bf16(A + arow * K, lds_off(&As[0][0]), K, M, K, 64, 64);
    tdm_load_2d_bf16(B + bcol * K, lds_off(&Bs[0][0]), K, N, K, 64, 128);
    __builtin_amdgcn_s_wait_tensorcnt(0);
  }
  __syncthreads();

  for (int s = 0; s < nchunks; ++s) {
    const int cur = s & 1;
    if (wave == 0 && (s + 1) < nchunks) {
      const int kk = (s + 1) << 6;
      tdm_load_2d_bf16(A + arow * K + kk, lds_off(&As[cur ^ 1][0]), K, M, K, 64, 64);
      tdm_load_2d_bf16(B + bcol * K + kk, lds_off(&Bs[cur ^ 1][0]), K, N, K, 64, 128);
    }
    const __bf16* a0p = &As[cur][(wm * 32 + r) * 64];
    const __bf16* a1p = a0p + 16 * 64;
    const __bf16* b0p = &Bs[cur][(wn * 32 + r) * 64];
    const __bf16* b1p = b0p + 16 * 64;
#pragma unroll
    for (int k2 = 0; k2 < 64; k2 += 32) {
      v16bf a0 = load_a_frag(a0p + k2, hi);
      v16bf a1 = load_a_frag(a1p + k2, hi);
      v16bf b0 = load_b_frag(b0p + k2, hi);
      v16bf b1 = load_b_frag(b1p + k2, hi);
      acc[0][0] = wmma_bf16(a0, b0, acc[0][0]);
      acc[0][1] = wmma_bf16(a0, b1, acc[0][1]);
      acc[1][0] = wmma_bf16(a1, b0, acc[1][0]);
      acc[1][1] = wmma_bf16(a1, b1, acc[1][1]);
    }
    if (wave == 0) __builtin_amdgcn_s_wait_tensorcnt(0);
    __syncthreads();
  }

  float* Cf = reinterpret_cast<float*>(Cout);
  __bf16* Cb = reinterpret_cast<__bf16*>(Cout);
  const long row0 = arow + wm * 32;
  const long col0 = bcol + wn * 32;
#pragma unroll
  for (int im = 0; im < 2; ++im) {
#pragma unroll
    for (int in_ = 0; in_ < 2; ++in_) {
      const long c = col0 + in_ * 16 + r;
#pragma unroll
      for (int v = 0; v < 8; ++v) {
        const long rr = row0 + im * 16 + v + 8 * hi;
        if (OUT_F32) Cf[rr * N + c] = acc[im][in_][v];
        else         Cb[rr * N + c] = (__bf16)acc[im][in_][v];
      }
    }
  }
}

// ---------------- Local windowed attention with XPos rotary -----------------
// One block per (bh, window). 256 threads = 8 waves; wave owns 16 query rows.
// qkv: [16384][2048] bf16 (cols 0..1023 = qk head-major, 1024..2047 = v)
// ctx: [16384][512] bf16 output (head-major)
__global__ __launch_bounds__(256) void attn_kernel(const __bf16* __restrict__ qkv,
                                                   __bf16* __restrict__ ctx) {
  __shared__ __bf16 qs[128 * 64];     // rope'd, scaled q   [row][dh]
  __shared__ __bf16 ks[128 * 64];     // rope'd k chunk     [key][dh]
  __shared__ __bf16 vT[64 * 128];     // v chunk transposed [dh][key]
  __shared__ __bf16 pst[8 * 16 * 32]; // per-wave p staging [wave][row16][key32]

  const int tid = threadIdx.x;
  const int lane = tid & 31, wave = tid >> 5;
  const int r = lane & 15, hi = lane >> 4;
  const int wq = blockIdx.x & 31;       // window
  const int bh = blockIdx.x >> 5;       // b*8 + h
  const int bidx = bh >> 3, head = bh & 7;
  const long tok0 = (long)bidx * 4096 + (long)wq * 128;

  // --- stage q with XPos rope (t = 256+i), * dh^-0.5 ---
  for (int e = tid; e < 128 * 64; e += 256) {
    const int i = e >> 6, d = e & 63, m = d & 31;
    const long base = (tok0 + i) * 2048 + head * 64;
    const float qv = (float)qkv[base + d];
    const float qo = (float)qkv[base + ((d < 32) ? d + 32 : d - 32)];
    const float rot = (d < 32) ? -qo : qo;
    const float invf = __expf(-(float)m * 0.2878231366242557f); // ln(1e4)/32
    const float tpos = (float)(256 + i);
    float sn, cs; __sincosf(tpos * invf, &sn, &cs);
    const float sb = ((float)(2 * m) + 25.6f) * (1.0f / 89.6f);
    const float sc = __expf(((tpos - 192.0f) * (1.0f / 64.0f)) * __logf(sb));
    qs[e] = (__bf16)(0.125f * (qv * cs + rot * sn) * sc);
  }
  __syncthreads();

  const v8f vz = {0.f, 0.f, 0.f, 0.f, 0.f, 0.f, 0.f, 0.f};
  v8f O[4] = {vz, vz, vz, vz};
  float mrow[8], srow[8];
#pragma unroll
  for (int v = 0; v < 8; ++v) { mrow[v] = NEGINF; srow[v] = 0.0f; }

  const v16bf aq0 = load_a_frag(&qs[(wave * 16 + r) * 64 + 0], hi);
  const v16bf aq1 = load_a_frag(&qs[(wave * 16 + r) * 64 + 32], hi);

  for (int c = 0; c < 3; ++c) {
    if ((wq == 0 && c == 0) || (wq == 31 && c == 2)) continue; // fully padded chunk
    __syncthreads(); // previous chunk consumed
    const long ktok0 = (long)bidx * 4096 + (long)(wq - 1 + c) * 128;
    // --- stage k chunk (rope, inverse scale) and v^T ---
    for (int e = tid; e < 128 * 64; e += 256) {
      const int i = e >> 6, d = e & 63, m = d & 31;
      const long base = (ktok0 + i) * 2048 + head * 64;
      const float kv = (float)qkv[base + d];
      const float ko = (float)qkv[base + ((d < 32) ? d + 32 : d - 32)];
      const float rot = (d < 32) ? -ko : ko;
      const float invf = __expf(-(float)m * 0.2878231366242557f);
      const float tpos = (float)(c * 128 + i);
      float sn, cs; __sincosf(tpos * invf, &sn, &cs);
      const float sb = ((float)(2 * m) + 25.6f) * (1.0f / 89.6f);
      const float isc = __expf(-((tpos - 192.0f) * (1.0f / 64.0f)) * __logf(sb));
      ks[e] = (__bf16)((kv * cs + rot * sn) * isc);
      vT[d * 128 + i] = qkv[base + 1024 + d];
    }
    __syncthreads();

    // --- compute: 4 groups of 32 keys ---
    for (int sub = 0; sub < 4; ++sub) {
      const int j0 = c * 128 + sub * 32;
      // sim tiles (16 queries x 16 keys each), K = dh = 64 -> 2 WMMAs per tile
      const __bf16* kr0 = &ks[(sub * 32 + r) * 64];
      const __bf16* kr1 = &ks[(sub * 32 + 16 + r) * 64];
      v8f t0 = vz, t1 = vz;
      t0 = wmma_bf16(aq0, load_b_frag(kr0, hi), t0);
      t0 = wmma_bf16(aq1, load_b_frag(kr0 + 32, hi), t0);
      t1 = wmma_bf16(aq0, load_b_frag(kr1, hi), t1);
      t1 = wmma_bf16(aq1, load_b_frag(kr1 + 32, hi), t1);

      // mask + online softmax update
#pragma unroll
      for (int v = 0; v < 8; ++v) {
        const int irow = wave * 16 + v + 8 * hi;
        const int ja = j0 + r, jb = j0 + 16 + r;
        float s0 = t0[v], s1 = t1[v];
        if (ja < irow || ja > irow + 256) s0 = NEGINF;
        if (jb < irow || jb > irow + 256) s1 = NEGINF;
        const float mt = hmax16(fmaxf(s0, s1));
        const float mnew = fmaxf(mrow[v], mt);
        const float alpha = (mrow[v] == NEGINF) ? 0.0f : __expf(mrow[v] - mnew);
        const float p0 = (s0 == NEGINF) ? 0.0f : __expf(s0 - mnew);
        const float p1 = (s1 == NEGINF) ? 0.0f : __expf(s1 - mnew);
        srow[v] = srow[v] * alpha + hsum16(p0 + p1);
        mrow[v] = mnew;
        O[0][v] *= alpha; O[1][v] *= alpha; O[2][v] *= alpha; O[3][v] *= alpha;
        pst[(wave * 16 + v + 8 * hi) * 32 + r]      = (__bf16)p0;
        pst[(wave * 16 + v + 8 * hi) * 32 + 16 + r] = (__bf16)p1;
      }
      // relayout p (accumulator layout -> A-frag) through wave-private LDS
      const v16bf pa = load_a_frag(&pst[(wave * 16 + r) * 32], hi);
      // O += p(16x32) @ v(32x64): 4 n-tiles, B-frag = contiguous row of v^T
#pragma unroll
      for (int nt = 0; nt < 4; ++nt) {
        const __bf16* vr = &vT[(nt * 16 + r) * 128 + sub * 32];
        O[nt] = wmma_bf16(pa, load_b_frag(vr, hi), O[nt]);
      }
    }
  }

  // --- finalize: O /= rowsum, write ctx head-major ---
#pragma unroll
  for (int v = 0; v < 8; ++v) {
    const float inv = 1.0f / srow[v];
    const long trow = tok0 + wave * 16 + v + 8 * hi;
#pragma unroll
    for (int nt = 0; nt < 4; ++nt)
      ctx[trow * 512 + head * 64 + nt * 16 + r] = (__bf16)(O[nt][v] * inv);
  }
}

// ---------------- launcher ----------------
extern "C" void kernel_launch(void* const* d_in, const int* in_sizes, int n_in,
                              void* d_out, int out_size, void* d_ws, size_t ws_size,
                              hipStream_t stream) {
  const float* x    = (const float*)d_in[0];
  const float* ln_g = (const float*)d_in[1];
  const float* ln_b = (const float*)d_in[2];
  const float* wqkv = (const float*)d_in[3];
  const float* wout = (const float*)d_in[4];

  char* ws = (char*)d_ws;
  __bf16* xn_b   = (__bf16*)(ws);                 //  32 MB : [16384][1024]
  __bf16* qkv_b  = (__bf16*)(ws + 33554432);      //  64 MB : [16384][2048]
  __bf16* ctx_b  = (__bf16*)(ws + 100663296);     //  16 MB : [16384][512]
  __bf16* wqkv_b = (__bf16*)(ws + 117440512);     //   4 MB : [2048][1024]
  __bf16* wout_b = (__bf16*)(ws + 121634816);     //   1 MB : [1024][512]

  cvt_bf16_kernel<<<2048, 256, 0, stream>>>(wqkv, wqkv_b, 2048L * 1024);
  cvt_bf16_kernel<<<512, 256, 0, stream>>>(wout, wout_b, 1024L * 512);
  ln_kernel<<<16384, 256, 0, stream>>>(x, ln_g, ln_b, xn_b);
  gemm_bf16_kernel<false><<<dim3(256, 16), 256, 0, stream>>>(xn_b, wqkv_b, qkv_b,
                                                             16384, 2048, 1024);
  attn_kernel<<<1024, 256, 0, stream>>>(qkv_b, ctx_b);
  gemm_bf16_kernel<true><<<dim3(256, 8), 256, 0, stream>>>(ctx_b, wout_b, d_out,
                                                           16384, 1024, 512);
}
